// MiMoV2Moe_3728031613183
// MI455X (gfx1250) — matrix operand order repaired and checked by reading, not compiled, not run on hardware
//
#include <hip/hip_runtime.h>
#include <hip/hip_bf16.h>

// ---------------------------------------------------------------------------
// MiMoV2 MoE (T=4096,H=2048,E=32,I=1024,K=8) for gfx1250 / MI455X.
// route -> group pairs by expert (32-aligned) -> grouped GEMMs with
// v_wmma_f32_16x16x32_bf16, 2 M-tiles per wave, LDS-staged A via
// global_load_async_to_lds_b128 + s_wait_asynccnt -> deterministic combine.
// Compute-bound (~4.1e11 FLOP vs ~60us HBM traffic) => bf16 WMMA path.
// ---------------------------------------------------------------------------

typedef __bf16 bf16;
typedef __attribute__((ext_vector_type(16))) __bf16 v16bf;
typedef __attribute__((ext_vector_type(8)))  __bf16 v8bf;
typedef __attribute__((ext_vector_type(8)))  float  v8f;
typedef int v4i_g __attribute__((vector_size(16)));   // GCC-style, matches builtin proto

#define T_NUM 4096
#define H_DIM 2048
#define E_NUM 32
#define I_DIM 1024
#define TOPK  8
#define CAP   (T_NUM * TOPK + E_NUM * 32)   // grouped rows incl. 32-pad per expert
#define KCHUNK 256                           // A-tile K staging chunk
#define LDS_STRIDE (KCHUNK + 8)              // +16B pad to stagger LDS banks

#if defined(__has_builtin)
#if __has_builtin(__builtin_amdgcn_global_load_async_to_lds_b128) && \
    __has_builtin(__builtin_amdgcn_s_wait_asynccnt)
#define USE_ASYNC_LDS 1
#endif
#endif
#ifndef USE_ASYNC_LDS
#define USE_ASYNC_LDS 0
#endif

#define AS1 __attribute__((address_space(1)))
#define AS3 __attribute__((address_space(3)))

// copy 16B global -> LDS (async DMA path when available)
__device__ __forceinline__ void lds_fill16(bf16* ldst, const bf16* gsrc) {
#if USE_ASYNC_LDS
    // LDS flat addr low 32 bits == LDS offset (aperture rule); AS1 == __device__
    AS3 v4i_g* l = (AS3 v4i_g*)(unsigned int)(unsigned long long)ldst;
    AS1 v4i_g* g = (AS1 v4i_g*)(unsigned long long)gsrc;
    __builtin_amdgcn_global_load_async_to_lds_b128(g, l, 0, 0);
#else
    *(v8bf*)ldst = *(const v8bf*)gsrc;
#endif
}

__device__ __forceinline__ void lds_fill_wait() {
#if USE_ASYNC_LDS
    __builtin_amdgcn_s_wait_asynccnt(0);
#endif
}

// --- A fragment for V_WMMA_F32_16X16X32_BF16 (contiguous K order used here, B loads match):
// lane m = lane&15 holds row m; hi = lane>>4 selects K ranges:
//   v0..v3 : K = hi*16 + 0..7      v4..v7 : K = hi*16 + 8..15
__device__ __forceinline__ v16bf load_a16_lds(const bf16* __restrict__ row, int kk, int hi) {
    v8bf lo  = *(const v8bf*)(row + kk + hi * 16);
    v8bf hi8 = *(const v8bf*)(row + kk + hi * 16 + 8);
    v16bf r;
#pragma unroll
    for (int i = 0; i < 8; ++i) { r[i] = lo[i]; r[i + 8] = hi8[i]; }
    return r;
}

// ------------------------- precision conversion ----------------------------
__global__ void __launch_bounds__(256) cvt_f32_bf16_kernel(const float* __restrict__ src,
                                                           bf16* __restrict__ dst, int n4) {
    int i = blockIdx.x * 256 + threadIdx.x;
    if (i >= n4) return;
    float4 v = ((const float4*)src)[i];
    bf16* d = dst + (size_t)i * 4;
    d[0] = (bf16)v.x; d[1] = (bf16)v.y; d[2] = (bf16)v.z; d[3] = (bf16)v.w;
}

// transpose+convert one [R,C] fp32 slice per expert into [C,R] bf16
__global__ void __launch_bounds__(256) tcvt_kernel(const float* __restrict__ src,
                                                   bf16* __restrict__ dst, int R, int C) {
    __shared__ float tile[32][33];
    int e = blockIdx.z;
    const float* s = src + (size_t)e * R * C;
    bf16* d = dst + (size_t)e * R * C;
    int c0 = blockIdx.x * 32, r0 = blockIdx.y * 32;
    int tx = threadIdx.x & 31, ty = threadIdx.x >> 5;   // 32 x 8
#pragma unroll
    for (int i = 0; i < 32; i += 8)
        tile[ty + i][tx] = s[(size_t)(r0 + ty + i) * C + (c0 + tx)];
    __syncthreads();
#pragma unroll
    for (int i = 0; i < 32; i += 8)
        d[(size_t)(c0 + ty + i) * R + (r0 + tx)] = (bf16)tile[tx][ty + i];
}

// ------------------------------ router -------------------------------------
// One wave32 per token; lane = expert. Softmax + greedy top-8 + renormalize.
__global__ void __launch_bounds__(32) router_kernel(const float* __restrict__ x,
                                                    const float* __restrict__ gw,
                                                    int* __restrict__ topk_ids,
                                                    float* __restrict__ pair_w,
                                                    int* __restrict__ counts) {
    int t = blockIdx.x;
    int lane = threadIdx.x;
    const float* xr = x + (size_t)t * H_DIM;
    float logit = 0.f;
    for (int h = 0; h < H_DIM; h += 4) {
        logit += xr[h + 0] * gw[(h + 0) * E_NUM + lane]
               + xr[h + 1] * gw[(h + 1) * E_NUM + lane]
               + xr[h + 2] * gw[(h + 2) * E_NUM + lane]
               + xr[h + 3] * gw[(h + 3) * E_NUM + lane];
    }
    float m = logit;
    for (int off = 16; off > 0; off >>= 1) m = fmaxf(m, __shfl_xor(m, off));
    float ex = __expf(logit - m);
    float s = ex;
    for (int off = 16; off > 0; off >>= 1) s += __shfl_xor(s, off);
    float score = ex / s;            // in (0,1]
    // greedy top-8, ties -> lowest expert index (matches jax.lax.top_k)
    float myw = 0.f; int myid = 0;
    float sc = score;
    for (int k = 0; k < TOPK; ++k) {
        float v = sc; int ie = lane;
        for (int off = 16; off > 0; off >>= 1) {
            float ov = __shfl_xor(v, off); int oi = __shfl_xor(ie, off);
            if (ov > v || (ov == v && oi < ie)) { v = ov; ie = oi; }
        }
        if (lane == k) { myw = v; myid = ie; }
        if (lane == ie) sc = -1.0f;
    }
    float wsum = 0.f;
    for (int j = 0; j < TOPK; ++j) wsum += __shfl(myw, j);
    if (lane < TOPK) {
        topk_ids[t * TOPK + lane] = myid;
        pair_w[t * TOPK + lane] = myw / wsum;
        atomicAdd(&counts[myid], 1);
    }
}

// ---------------- expert grouping: scan + pad + scatter --------------------
__global__ void __launch_bounds__(32) scan_kernel(const int* __restrict__ counts,
                                                  int* __restrict__ offsets,
                                                  int* __restrict__ counts_pad,
                                                  int* __restrict__ grouped_tok,
                                                  int* __restrict__ grouped_pair) {
    __shared__ int soff[E_NUM];
    int e = threadIdx.x;
    if (e == 0) {
        int acc = 0;
        for (int j = 0; j < E_NUM; ++j) {
            soff[j] = acc;
            acc += (counts[j] + 31) & ~31;
        }
    }
    __syncthreads();
    int c = counts[e];
    int pad = (c + 31) & ~31;
    offsets[e] = soff[e];
    counts_pad[e] = pad;
    for (int i = c; i < pad; ++i) {       // pad rows: token 0, no output pair
        grouped_tok[soff[e] + i] = 0;
        grouped_pair[soff[e] + i] = -1;
    }
}

__global__ void __launch_bounds__(256) build_kernel(const int* __restrict__ topk_ids,
                                                    int* __restrict__ cursor,
                                                    const int* __restrict__ offsets,
                                                    int* __restrict__ grouped_tok,
                                                    int* __restrict__ grouped_pair) {
    int p = blockIdx.x * 256 + threadIdx.x;
    if (p >= T_NUM * TOPK) return;
    int e = topk_ids[p];
    int slot = atomicAdd(&cursor[e], 1);
    int r = offsets[e] + slot;
    grouped_tok[r] = p >> 3;   // token id
    grouped_pair[r] = p;       // (t,k) pair id
}

// ---------------- grouped GEMM 1: fused gate+up+SiLU -----------------------
// Block = 8 wave32s, 32 rows x 128 cols. Wave owns 2 M-tiles x 1 N-tile:
// each B fragment pair feeds 4 WMMAs. A staged in LDS per 256-wide K chunk.
__global__ void __launch_bounds__(256) moe_gateup_kernel(const bf16* __restrict__ xb,
                                                         const bf16* __restrict__ wgT,
                                                         const bf16* __restrict__ wuT,
                                                         const int* __restrict__ grouped_tok,
                                                         const int* __restrict__ counts_pad,
                                                         const int* __restrict__ offsets,
                                                         bf16* __restrict__ hbuf) {
    __shared__ bf16 tile[32][LDS_STRIDE];
    int e = blockIdx.z;
    int padded = counts_pad[e];
    if ((int)blockIdx.y * 32 >= padded) return;      // dynamic row count, early exit
    int rbase = offsets[e] + blockIdx.y * 32;
    int wave = threadIdx.x >> 5, lane = threadIdx.x & 31;
    int n = lane & 15, hi = lane >> 4;
    int ncol = blockIdx.x * 128 + wave * 16 + n;

    // staging assignment: thread -> (row, 64B segment)
    int frow = threadIdx.x >> 3;                     // 0..31
    int fseg = threadIdx.x & 7;                      // 8 segs of 32 bf16
    const bf16* fill_row = xb + (size_t)grouped_tok[rbase + frow] * H_DIM + fseg * 32;

    const bf16* bg = wgT + ((size_t)e * I_DIM + ncol) * H_DIM;   // [E,I,H] : col-contig
    const bf16* bu = wuT + ((size_t)e * I_DIM + ncol) * H_DIM;

    v8f accg0 = {}, accu0 = {}, accg1 = {}, accu1 = {};
    int m0 = n, m1 = 16 + n;

    for (int kc = 0; kc < H_DIM; kc += KCHUNK) {
        // stage A[32 x KCHUNK] into LDS (4 x 16B per thread)
#pragma unroll
        for (int i = 0; i < 4; ++i)
            lds_fill16(&tile[frow][fseg * 32 + i * 8], fill_row + kc + i * 8);
        lds_fill_wait();
        __syncthreads();

#pragma unroll
        for (int kk = 0; kk < KCHUNK; kk += 32) {
            int k = kc + kk;
            v16bf a0 = load_a16_lds(&tile[m0][0], kk, hi);
            v16bf a1 = load_a16_lds(&tile[m1][0], kk, hi);
            v16bf bgf = *(const v16bf*)(bg + k + hi * 16);   // lanes0-15: K=k..k+15
            v16bf buf2 = *(const v16bf*)(bu + k + hi * 16);  // lanes16-31: K=k+16..k+31
            __builtin_prefetch(bg + k + 512, 0, 1);
            __builtin_prefetch(bu + k + 512, 0, 1);
            accg0 = __builtin_amdgcn_wmma_f32_16x16x32_bf16(false, a0, false, bgf,
                                                            (short)0, accg0, false, false);
            accu0 = __builtin_amdgcn_wmma_f32_16x16x32_bf16(false, a0, false, buf2,
                                                            (short)0, accu0, false, false);
            accg1 = __builtin_amdgcn_wmma_f32_16x16x32_bf16(false, a1, false, bgf,
                                                            (short)0, accg1, false, false);
            accu1 = __builtin_amdgcn_wmma_f32_16x16x32_bf16(false, a1, false, buf2,
                                                            (short)0, accu1, false, false);
        }
        __syncthreads();
    }
    // C/D layout: VGPR v -> row hi*8+v, col = lane&15
#pragma unroll
    for (int v = 0; v < 8; ++v) {
        int mrow = hi * 8 + v;
        float g0 = accg0[v], u0 = accu0[v];
        float g1 = accg1[v], u1 = accu1[v];
        hbuf[(size_t)(rbase + mrow) * I_DIM + ncol] = (bf16)((g0 / (1.0f + __expf(-g0))) * u0);
        hbuf[(size_t)(rbase + 16 + mrow) * I_DIM + ncol] = (bf16)((g1 / (1.0f + __expf(-g1))) * u1);
    }
}

// ---------------- grouped GEMM 2: down projection --------------------------
__global__ void __launch_bounds__(256) moe_down_kernel(const bf16* __restrict__ hbuf,
                                                       const bf16* __restrict__ wdT,
                                                       const int* __restrict__ grouped_pair,
                                                       const int* __restrict__ counts_pad,
                                                       const int* __restrict__ offsets,
                                                       bf16* __restrict__ pair_out) {
    __shared__ bf16 tile[32][LDS_STRIDE];
    int e = blockIdx.z;
    int padded = counts_pad[e];
    if ((int)blockIdx.y * 32 >= padded) return;
    int rbase = offsets[e] + blockIdx.y * 32;
    int wave = threadIdx.x >> 5, lane = threadIdx.x & 31;
    int n = lane & 15, hi = lane >> 4;
    int ncol = blockIdx.x * 128 + wave * 16 + n;

    int frow = threadIdx.x >> 3;
    int fseg = threadIdx.x & 7;
    const bf16* fill_row = hbuf + (size_t)(rbase + frow) * I_DIM + fseg * 32;

    const bf16* bd = wdT + ((size_t)e * H_DIM + ncol) * I_DIM;   // [E,H,I] : col-contig

    v8f acc0 = {}, acc1 = {};
    int m0 = n, m1 = 16 + n;

    for (int kc = 0; kc < I_DIM; kc += KCHUNK) {
#pragma unroll
        for (int i = 0; i < 4; ++i)
            lds_fill16(&tile[frow][fseg * 32 + i * 8], fill_row + kc + i * 8);
        lds_fill_wait();
        __syncthreads();

#pragma unroll
        for (int kk = 0; kk < KCHUNK; kk += 32) {
            int k = kc + kk;
            v16bf a0 = load_a16_lds(&tile[m0][0], kk, hi);
            v16bf a1 = load_a16_lds(&tile[m1][0], kk, hi);
            v16bf b = *(const v16bf*)(bd + k + hi * 16);
            __builtin_prefetch(bd + k + 512, 0, 1);
            acc0 = __builtin_amdgcn_wmma_f32_16x16x32_bf16(false, a0, false, b,
                                                           (short)0, acc0, false, false);
            acc1 = __builtin_amdgcn_wmma_f32_16x16x32_bf16(false, a1, false, b,
                                                           (short)0, acc1, false, false);
        }
        __syncthreads();
    }
#pragma unroll
    for (int v = 0; v < 8; ++v) {
        int mrow = hi * 8 + v;
        int p0 = grouped_pair[rbase + mrow];
        int p1 = grouped_pair[rbase + 16 + mrow];
        if (p0 >= 0) pair_out[(size_t)p0 * H_DIM + ncol] = (bf16)acc0[v];
        if (p1 >= 0) pair_out[(size_t)p1 * H_DIM + ncol] = (bf16)acc1[v];
    }
}

// ---------------- deterministic per-token combine --------------------------
__global__ void __launch_bounds__(256) combine_kernel(const bf16* __restrict__ pair_out,
                                                      const float* __restrict__ pair_w,
                                                      float* __restrict__ out) {
    int t = blockIdx.x;
    float w[TOPK];
#pragma unroll
    for (int k = 0; k < TOPK; ++k) w[k] = pair_w[t * TOPK + k];
    for (int h = threadIdx.x; h < H_DIM; h += 256) {
        float acc = 0.f;
#pragma unroll
        for (int k = 0; k < TOPK; ++k)
            acc += w[k] * (float)pair_out[(size_t)(t * TOPK + k) * H_DIM + h];
        out[(size_t)t * H_DIM + h] = acc;
    }
}

// ---------------------------------------------------------------------------
extern "C" void kernel_launch(void* const* d_in, const int* in_sizes, int n_in,
                              void* d_out, int out_size, void* d_ws, size_t ws_size,
                              hipStream_t stream) {
    const float* x  = (const float*)d_in[0];   // [T,H]
    const float* gw = (const float*)d_in[1];   // [H,E]
    const float* wg = (const float*)d_in[2];   // [E,H,I]
    const float* wu = (const float*)d_in[3];   // [E,H,I]
    const float* wd = (const float*)d_in[4];   // [E,I,H]

    float* out = (float*)d_out;                           // [T,H] fp32
    int* topk_ids = (int*)(out + (size_t)T_NUM * H_DIM);  // [T,8] int32 tail

    // workspace partition (~623 MB total)
    char* ws = (char*)d_ws;
    size_t off = 0;
    auto alloc = [&](size_t bytes) -> void* {
        size_t a = (off + 255) & ~(size_t)255;
        off = a + bytes;
        return (void*)(ws + a);
    };
    bf16* xb        = (bf16*)alloc((size_t)T_NUM * H_DIM * 2);
    bf16* wgT       = (bf16*)alloc((size_t)E_NUM * I_DIM * H_DIM * 2);
    bf16* wuT       = (bf16*)alloc((size_t)E_NUM * I_DIM * H_DIM * 2);
    bf16* wdT       = (bf16*)alloc((size_t)E_NUM * H_DIM * I_DIM * 2);
    float* pair_w   = (float*)alloc((size_t)T_NUM * TOPK * 4);
    int* counts     = (int*)alloc(E_NUM * 4);
    int* cursor     = (int*)alloc(E_NUM * 4);
    int* offsets    = (int*)alloc(E_NUM * 4);
    int* counts_pad = (int*)alloc(E_NUM * 4);
    int* gtok       = (int*)alloc((size_t)CAP * 4);
    int* gpair      = (int*)alloc((size_t)CAP * 4);
    bf16* hbuf      = (bf16*)alloc((size_t)CAP * I_DIM * 2);
    bf16* pair_out  = (bf16*)alloc((size_t)T_NUM * TOPK * H_DIM * 2);
    (void)ws_size; (void)in_sizes; (void)n_in; (void)out_size;

    (void)hipMemsetAsync(counts, 0, E_NUM * 4, stream);
    (void)hipMemsetAsync(cursor, 0, E_NUM * 4, stream);

    // 1. activations -> bf16
    {
        int n4 = (T_NUM * H_DIM) / 4;
        cvt_f32_bf16_kernel<<<(n4 + 255) / 256, 256, 0, stream>>>(x, xb, n4);
    }
    // 2. weights -> bf16, transposed so WMMA B-columns are contiguous
    tcvt_kernel<<<dim3(I_DIM / 32, H_DIM / 32, E_NUM), 256, 0, stream>>>(wg, wgT, H_DIM, I_DIM);
    tcvt_kernel<<<dim3(I_DIM / 32, H_DIM / 32, E_NUM), 256, 0, stream>>>(wu, wuT, H_DIM, I_DIM);
    tcvt_kernel<<<dim3(H_DIM / 32, I_DIM / 32, E_NUM), 256, 0, stream>>>(wd, wdT, I_DIM, H_DIM);
    // 3. routing (softmax + top-8 + renorm), counts
    router_kernel<<<T_NUM, 32, 0, stream>>>(x, gw, topk_ids, pair_w, counts);
    // 4. group pairs by expert (32-aligned regions)
    scan_kernel<<<1, 32, 0, stream>>>(counts, offsets, counts_pad, gtok, gpair);
    build_kernel<<<(T_NUM * TOPK + 255) / 256, 256, 0, stream>>>(topk_ids, cursor, offsets,
                                                                 gtok, gpair);
    // 5. grouped fused gate/up GEMM + SiLU (WMMA bf16, LDS-staged A)
    moe_gateup_kernel<<<dim3(I_DIM / 128, T_NUM / 32, E_NUM), 256, 0, stream>>>(
        xb, wgT, wuT, gtok, counts_pad, offsets, hbuf);
    // 6. grouped down GEMM (WMMA bf16, LDS-staged A)
    moe_down_kernel<<<dim3(H_DIM / 128, T_NUM / 32, E_NUM), 256, 0, stream>>>(
        hbuf, wdT, gpair, counts_pad, offsets, pair_out);
    // 7. deterministic weighted combine
    combine_kernel<<<T_NUM, 256, 0, stream>>>(pair_out, pair_w, out);
}
